// EvaLinearAttention_20890720928474
// MI455X (gfx1250) — matrix operand order repaired and checked by reading, not compiled, not run on hardware
//
#include <hip/hip_runtime.h>
#include <hip/hip_bf16.h>

// ---------------------------------------------------------------------------
// EVA linear attention for MI455X (gfx1250, wave32, WMMA bf16 16x16x32)
// B=8 N=4096 C=768 H=12 D=64 npt=1
// ---------------------------------------------------------------------------

typedef __attribute__((ext_vector_type(16))) __bf16 bf16x16;
typedef __attribute__((ext_vector_type(8)))  float  f32x8;

union FragU { uint4 q[2]; bf16x16 v; };
union U4    { uint4 q; unsigned int u[4]; unsigned short s[8]; };

__device__ __forceinline__ unsigned short f2bf(float x) {
    unsigned int u = __builtin_bit_cast(unsigned int, x);
    unsigned int r = u + 0x7FFFu + ((u >> 16) & 1u);   // round-to-nearest-even
    return (unsigned short)(r >> 16);
}
__device__ __forceinline__ float bf2f(unsigned int h16) {
    unsigned int u = (h16 & 0xFFFFu) << 16;
    return __builtin_bit_cast(float, u);
}
__device__ __forceinline__ unsigned int pack2(float lo, float hi) {
    return (unsigned int)f2bf(lo) | ((unsigned int)f2bf(hi) << 16);
}

// A fragment (16x32 bf16, M x K): lane holds row M=lane&15.
// half=0 lanes: K {0..7} and {16..23};  half=1 lanes: K {8..15} and {24..31}.
__device__ __forceinline__ bf16x16 fragA(const unsigned short* rowp, int half) {
    FragU f;
    f.q[0] = *(const uint4*)(rowp + (half ? 8  : 0));
    f.q[1] = *(const uint4*)(rowp + (half ? 24 : 16));
    return f.v;
}
// B fragment (32x16 bf16, K x N), column-major staged (colp -> 32 contiguous K):
// half=0 lanes: K {0..7} and {16..23};  half=1 lanes: K {8..15} and {24..31} (col = lane&15).
__device__ __forceinline__ bf16x16 fragB(const unsigned short* colp, int half) {
    FragU f;
    f.q[0] = *(const uint4*)(colp + (half ? 8  : 0));
    f.q[1] = *(const uint4*)(colp + (half ? 24 : 16));
    return f.v;
}

__device__ __forceinline__ f32x8 wmma_bf16(bf16x16 a, bf16x16 b, f32x8 c) {
    return __builtin_amdgcn_wmma_f32_16x16x32_bf16(false, a, false, b,
                                                   (short)0, c, false, false);
}

// ---------------------------------------------------------------------------
// Generic WMMA GEMM: Out[M x Nc] = A[M x K] @ W^T + bias,  W is [Nc x K] f32.
// Tile 128x128x32, 256 threads = 8 waves (2x4), each wave 4x2 16x16 tiles.
// Double-buffered LDS (one barrier per K-step), packed b128 LDS stores.
// ---------------------------------------------------------------------------
template<bool A_F32, bool OUT_BF16>
__global__ __launch_bounds__(256) void gemm_wmma(
    const void* __restrict__ Av, const float* __restrict__ W,
    const float* __restrict__ bias, void* __restrict__ Ov,
    int M, int Nc, int K)
{
    __shared__ unsigned int ldsA32[2][128 * 16];   // [m][k] pairs
    __shared__ unsigned int ldsB32[2][128 * 16];   // [n][k] pairs (col-major B)

    const int nTilesN = Nc >> 7;
    const int bm = blockIdx.x / nTilesN;
    const int bn = blockIdx.x % nTilesN;
    const int m0 = bm << 7, n0 = bn << 7;

    const int t    = threadIdx.x;
    const int lane = t & 31, wave = t >> 5;
    const int half = lane >> 4, lr = lane & 15;
    const int wrow = (wave >> 2) * 64;     // 0 or 64
    const int wcol = (wave & 3) * 32;      // 0,32,64,96

    f32x8 acc[4][2];
#pragma unroll
    for (int mi = 0; mi < 4; ++mi)
#pragma unroll
        for (int ni = 0; ni < 2; ++ni)
#pragma unroll
            for (int i = 0; i < 8; ++i) acc[mi][ni][i] = 0.0f;

    const int lrow = t >> 1;            // 0..127
    const int lkh  = (t & 1) << 4;      // 0 or 16
    const int luo  = lrow * 16 + (lkh >> 1);    // uint offset in tile

    auto stage = [&](int buf, int k0) {
        // ---- A tile ----
        if (A_F32) {
            const float* src = (const float*)Av + (size_t)(m0 + lrow) * K + k0 + lkh;
            unsigned int w_[8];
#pragma unroll
            for (int i = 0; i < 4; ++i) {
                float4 f = *(const float4*)(src + 4 * i);
                w_[2 * i]     = pack2(f.x, f.y);
                w_[2 * i + 1] = pack2(f.z, f.w);
            }
            uint4* d = (uint4*)&ldsA32[buf][luo];
            uint4 q0; q0.x = w_[0]; q0.y = w_[1]; q0.z = w_[2]; q0.w = w_[3];
            uint4 q1; q1.x = w_[4]; q1.y = w_[5]; q1.z = w_[6]; q1.w = w_[7];
            d[0] = q0; d[1] = q1;
        } else {
            const uint4* s = (const uint4*)((const unsigned short*)Av +
                                            (size_t)(m0 + lrow) * K + k0 + lkh);
            uint4* d = (uint4*)&ldsA32[buf][luo];
            d[0] = s[0]; d[1] = s[1];
        }
        // ---- B tile: B[k][n] = W[n][k]; W rows contiguous in k ----
        {
            const float* src = W + (size_t)(n0 + lrow) * K + k0 + lkh;
            unsigned int w_[8];
#pragma unroll
            for (int i = 0; i < 4; ++i) {
                float4 f = *(const float4*)(src + 4 * i);
                w_[2 * i]     = pack2(f.x, f.y);
                w_[2 * i + 1] = pack2(f.z, f.w);
            }
            uint4* d = (uint4*)&ldsB32[buf][luo];
            uint4 q0; q0.x = w_[0]; q0.y = w_[1]; q0.z = w_[2]; q0.w = w_[3];
            uint4 q1; q1.x = w_[4]; q1.y = w_[5]; q1.z = w_[6]; q1.w = w_[7];
            d[0] = q0; d[1] = q1;
        }
    };

    const int steps = K >> 5;
    stage(0, 0);
    __syncthreads();

    for (int kt = 0; kt < steps; ++kt) {
        if (kt + 1 < steps) stage((kt + 1) & 1, (kt + 1) << 5);

        const unsigned short* sa = (const unsigned short*)ldsA32[kt & 1];
        const unsigned short* sb = (const unsigned short*)ldsB32[kt & 1];

        bf16x16 af[4], bfm[2];
#pragma unroll
        for (int mi = 0; mi < 4; ++mi)
            af[mi] = fragA(sa + (wrow + mi * 16 + lr) * 32, half);
#pragma unroll
        for (int ni = 0; ni < 2; ++ni)
            bfm[ni] = fragB(sb + (wcol + ni * 16 + lr) * 32, half);
#pragma unroll
        for (int mi = 0; mi < 4; ++mi)
#pragma unroll
            for (int ni = 0; ni < 2; ++ni)
                acc[mi][ni] = wmma_bf16(af[mi], bfm[ni], acc[mi][ni]);
        __syncthreads();
    }

    // ---- epilogue: bias add, store ----
#pragma unroll
    for (int mi = 0; mi < 4; ++mi)
#pragma unroll
        for (int ni = 0; ni < 2; ++ni) {
            const int colg = n0 + wcol + ni * 16 + lr;
            const float bv = bias[colg];
#pragma unroll
            for (int r = 0; r < 8; ++r) {
                const int rowg = m0 + wrow + mi * 16 + r + (half ? 8 : 0);
                const float v = acc[mi][ni][r] + bv;
                if (OUT_BF16)
                    ((unsigned short*)Ov)[(size_t)rowg * Nc + colg] = f2bf(v);
                else
                    ((float*)Ov)[(size_t)rowg * Nc + colg] = v;
            }
        }
}

// ---------------------------------------------------------------------------
// RoPE (interleaved) + softmax over D=64, in place on q/k slices of qkv (bf16).
// One wave per (row, head, which), 2 elements per lane (rope pair in-lane).
// ---------------------------------------------------------------------------
__global__ __launch_bounds__(256) void rope_softmax_kernel(
    unsigned short* __restrict__ qkv, const float* __restrict__ rope)
{
    const int gw   = blockIdx.x * 8 + (threadIdx.x >> 5);   // over B*N*H*2
    const int lane = threadIdx.x & 31;
    const int which = gw & 1;            // 0=q, 1=k
    const int h     = (gw >> 1) % 12;
    const int row   = gw / 24;           // b*4096+n
    const int n     = row & 4095;

    const size_t base = (size_t)row * 2304 + which * 768 + h * 64;
    const int d0 = lane * 2;

    const unsigned int w0 = *(const unsigned int*)(qkv + base + d0);
    float t0 = bf2f(w0), t1 = bf2f(w0 >> 16);

    if (n >= 1) {
        const float* rp = rope + (size_t)(n - 1) * 128;
        const float s0 = rp[d0], s1 = rp[d0 + 1];
        const float c0 = rp[64 + d0], c1 = rp[64 + d0 + 1];
        const float o0 = t0 * c0 - t1 * s0;     // rot[2i]   = -t[2i+1]
        const float o1 = t1 * c1 + t0 * s1;     // rot[2i+1] =  t[2i]
        t0 = o0; t1 = o1;
    }
    float m = fmaxf(t0, t1);
#pragma unroll
    for (int off = 16; off > 0; off >>= 1) m = fmaxf(m, __shfl_xor(m, off, 32));
    const float e0 = __expf(t0 - m), e1 = __expf(t1 - m);
    float s = e0 + e1;
#pragma unroll
    for (int off = 16; off > 0; off >>= 1) s += __shfl_xor(s, off, 32);
    const float inv = 1.0f / s;
    *(unsigned int*)(qkv + base + d0) = pack2(e0 * inv, e1 * inv);
}

// ksum[bh][d] = sum_n phi_k[b,n,h,d]   (packed dword loads, 2 d per thread)
__global__ __launch_bounds__(256) void ksum_kernel(
    const unsigned short* __restrict__ qkv, float* __restrict__ ksum)
{
    const int bh = blockIdx.x, b = bh / 12, h = bh % 12;
    const int t = threadIdx.x, dp = t & 31, ch = t >> 5;   // 8 n-chunks
    float s0 = 0.0f, s1 = 0.0f;
    for (int n = ch; n < 4096; n += 8) {
        const unsigned int w =
            *(const unsigned int*)(qkv + (size_t)(b * 4096 + n) * 2304 + 768 +
                                   h * 64 + 2 * dp);
        s0 += bf2f(w);
        s1 += bf2f(w >> 16);
    }
    __shared__ float red0[256], red1[256];
    red0[t] = s0; red1[t] = s1;
    __syncthreads();
    if (ch == 0) {
        float a = 0.0f, c = 0.0f;
#pragma unroll
        for (int i = 0; i < 8; ++i) { a += red0[i * 32 + dp]; c += red1[i * 32 + dp]; }
        ksum[bh * 64 + 2 * dp]     = a;
        ksum[bh * 64 + 2 * dp + 1] = c;
    }
}

// z[bh][n] = dot(phi_q[b,n,h,:], ksum[bh,:])
__global__ __launch_bounds__(256) void z_kernel(
    const unsigned short* __restrict__ qkv, const float* __restrict__ ksum,
    float* __restrict__ z)
{
    const int gw   = blockIdx.x * 8 + (threadIdx.x >> 5);   // over B*N*H
    const int lane = threadIdx.x & 31;
    const int h = gw % 12, row = gw / 12;
    const int b = row >> 12, n = row & 4095;
    const unsigned int w =
        *(const unsigned int*)(qkv + (size_t)row * 2304 + h * 64 + lane * 2);
    const float* ks = ksum + (b * 12 + h) * 64 + lane * 2;
    float s = bf2f(w) * ks[0] + bf2f(w >> 16) * ks[1];
#pragma unroll
    for (int off = 16; off > 0; off >>= 1) s += __shfl_xor(s, off, 32);
    if (lane == 0) z[(size_t)(b * 12 + h) * 4096 + n] = s;
}

// kv[bh] (64x64) = phi_k^T @ v ; one block (4 waves) per (b,h), K=N=4096.
// Transposed staging packs (k, k+1) pairs so the scatter is b32 stores.
__global__ __launch_bounds__(128) void kv_kernel(
    const unsigned short* __restrict__ qkv, float* __restrict__ kv)
{
    __shared__ unsigned int ldsA32[64 * 16];  // [d][k/2]  = phi_k^T tile
    __shared__ unsigned int ldsB32[64 * 16];  // [e][k/2]  = v tile (col-major)
    const int bh = blockIdx.x, b = bh / 12, h = bh % 12;
    const int t = threadIdx.x, lane = t & 31, wave = t >> 5;
    const int half = lane >> 4, lr = lane & 15;

    f32x8 acc[4];
#pragma unroll
    for (int ni = 0; ni < 4; ++ni)
#pragma unroll
        for (int i = 0; i < 8; ++i) acc[ni][i] = 0.0f;

    const int pair = t & 15;       // k-pair: rows n0+2*pair, n0+2*pair+1
    const int dch  = t >> 4;       // 0..7 : d chunk of 8

    for (int n0 = 0; n0 < 4096; n0 += 32) {
        const unsigned short* r0 =
            qkv + (size_t)(b * 4096 + n0 + 2 * pair) * 2304 + h * 64 + dch * 8;
        const unsigned short* r1 = r0 + 2304;
        U4 ka, kb, va, vb;
        ka.q = *(const uint4*)(r0 + 768);
        kb.q = *(const uint4*)(r1 + 768);
        va.q = *(const uint4*)(r0 + 1536);
        vb.q = *(const uint4*)(r1 + 1536);
        if (n0 + 32 < 4096) {                      // stream-ahead hint
            __builtin_prefetch(r0 + 32 * 2304 + 768, 0, 0);
            __builtin_prefetch(r0 + 32 * 2304 + 1536, 0, 0);
        }
        __syncthreads();
#pragma unroll
        for (int i = 0; i < 8; ++i) {
            const int d = dch * 8 + i;
            ldsA32[d * 16 + pair] = (unsigned)ka.s[i] | ((unsigned)kb.s[i] << 16);
            ldsB32[d * 16 + pair] = (unsigned)va.s[i] | ((unsigned)vb.s[i] << 16);
        }
        __syncthreads();

        const unsigned short* sa = (const unsigned short*)ldsA32;
        const unsigned short* sb = (const unsigned short*)ldsB32;
        bf16x16 a = fragA(sa + (wave * 16 + lr) * 32, half);
#pragma unroll
        for (int ni = 0; ni < 4; ++ni) {
            bf16x16 bb = fragB(sb + (ni * 16 + lr) * 32, half);
            acc[ni] = wmma_bf16(a, bb, acc[ni]);
        }
    }
    float* o = kv + (size_t)bh * 4096;
#pragma unroll
    for (int ni = 0; ni < 4; ++ni)
#pragma unroll
        for (int r = 0; r < 8; ++r) {
            const int m = wave * 16 + r + (half ? 8 : 0);
            o[m * 64 + ni * 16 + lr] = acc[ni][r];
        }
}

// out = (phi_q @ kv) / (z + eps); heads re-interleaved into O[B*N][C] bf16.
__global__ __launch_bounds__(256) void attn_out_kernel(
    const unsigned short* __restrict__ qkv, const float* __restrict__ kv,
    const float* __restrict__ z, unsigned short* __restrict__ O)
{
    __shared__ unsigned int ldsKV32[64 * 32];   // [e][d/2] bf16 (kv transposed)
    const int blk = blockIdx.x;
    const int nb = blk & 31;            // 4096/128
    const int bh = blk >> 5;
    const int b = bh / 12, h = bh % 12;
    const int t = threadIdx.x, lane = t & 31, wave = t >> 5;
    const int half = lane >> 4, lr = lane & 15;

    {   // stage kv -> LDS, transposed + bf16, packed (d, d+1) pairs
        const int dp = t & 31, ech = t >> 5;      // d pair, e chunk of 8
        const float* kr0 = kv + (size_t)bh * 4096 + (2 * dp) * 64 + ech * 8;
        const float* kr1 = kr0 + 64;
        float4 a0 = *(const float4*)kr0, a1 = *(const float4*)(kr0 + 4);
        float4 b0 = *(const float4*)kr1, b1 = *(const float4*)(kr1 + 4);
        const float lo[8] = {a0.x, a0.y, a0.z, a0.w, a1.x, a1.y, a1.z, a1.w};
        const float hi[8] = {b0.x, b0.y, b0.z, b0.w, b1.x, b1.y, b1.z, b1.w};
#pragma unroll
        for (int i = 0; i < 8; ++i) {
            const int e = ech * 8 + i;
            ldsKV32[e * 32 + dp] = pack2(lo[i], hi[i]);
        }
    }
    __syncthreads();

    const int n0 = nb * 128 + wave * 16;
    f32x8 acc[4];
#pragma unroll
    for (int ni = 0; ni < 4; ++ni)
#pragma unroll
        for (int i = 0; i < 8; ++i) acc[ni][i] = 0.0f;

    const unsigned short* ldsKV = (const unsigned short*)ldsKV32;
    const size_t arow = (size_t)(b * 4096 + n0 + lr) * 2304 + h * 64;
#pragma unroll
    for (int k0 = 0; k0 < 64; k0 += 32) {
        FragU f;
        const unsigned short* p = qkv + arow + k0;
        f.q[0] = *(const uint4*)(p + (half ? 8  : 0));
        f.q[1] = *(const uint4*)(p + (half ? 24 : 16));
        const bf16x16 a = f.v;
#pragma unroll
        for (int ni = 0; ni < 4; ++ni) {
            bf16x16 bb = fragB(ldsKV + (ni * 16 + lr) * 64 + k0, half);
            acc[ni] = wmma_bf16(a, bb, acc[ni]);
        }
    }

    const float* zp = z + (size_t)bh * 4096;
#pragma unroll
    for (int ni = 0; ni < 4; ++ni)
#pragma unroll
        for (int r = 0; r < 8; ++r) {
            const int n = n0 + r + (half ? 8 : 0);
            const float zz = zp[n] + 1e-5f;
            const float v = acc[ni][r] / zz;
            O[(size_t)(b * 4096 + n) * 768 + h * 64 + ni * 16 + lr] = f2bf(v);
        }
}

// LayerNorm over C=768, in place on bf16 O. One block per row (coalesced cols).
__global__ __launch_bounds__(256) void ln_kernel(
    unsigned short* __restrict__ O, const float* __restrict__ g,
    const float* __restrict__ bb)
{
    const int row = blockIdx.x, t = threadIdx.x;
    unsigned short* p = O + (size_t)row * 768;
    float x[3], s = 0.0f, s2 = 0.0f;
#pragma unroll
    for (int i = 0; i < 3; ++i) {
        x[i] = bf2f(p[t + i * 256]);
        s += x[i];
        s2 += x[i] * x[i];
    }
    __shared__ float rs[256], rs2[256];
    rs[t] = s; rs2[t] = s2;
    __syncthreads();
    for (int off = 128; off > 0; off >>= 1) {
        if (t < off) { rs[t] += rs[t + off]; rs2[t] += rs2[t + off]; }
        __syncthreads();
    }
    const float mean = rs[0] * (1.0f / 768.0f);
    const float var  = rs2[0] * (1.0f / 768.0f) - mean * mean;
    const float inv  = rsqrtf(var + 1e-5f);
#pragma unroll
    for (int i = 0; i < 3; ++i) {
        const int c = t + i * 256;
        p[c] = f2bf((x[i] - mean) * inv * g[c] + bb[c]);
    }
}

__global__ void pack_bias_kernel(const float* __restrict__ qb,
                                 const float* __restrict__ vb,
                                 float* __restrict__ out)
{
    const int i = blockIdx.x * 256 + threadIdx.x;
    if (i < 2304)
        out[i] = (i < 768) ? qb[i] : ((i < 1536) ? 0.0f : vb[i - 1536]);
}

// ---------------------------------------------------------------------------
extern "C" void kernel_launch(void* const* d_in, const int* in_sizes, int n_in,
                              void* d_out, int out_size, void* d_ws, size_t ws_size,
                              hipStream_t stream)
{
    const float* x      = (const float*)d_in[0];
    const float* rope   = (const float*)d_in[1];
    const float* qkv_w  = (const float*)d_in[2];
    const float* q_bias = (const float*)d_in[3];
    const float* v_bias = (const float*)d_in[4];
    const float* norm_g = (const float*)d_in[5];
    const float* norm_b = (const float*)d_in[6];
    const float* proj_w = (const float*)d_in[7];
    const float* proj_b = (const float*)d_in[8];
    float* out = (float*)d_out;

    const int M = 8 * 4096;            // 32768 rows

    char* w = (char*)d_ws;
    auto carve = [&](size_t bytes) {
        char* p = w;
        w += (bytes + 255) & ~(size_t)255;
        return p;
    };
    float*          qkvbias = (float*)carve(2304 * sizeof(float));
    unsigned short* qkv     = (unsigned short*)carve((size_t)M * 2304 * 2);
    unsigned short* O       = (unsigned short*)carve((size_t)M * 768 * 2);
    float*          kv      = (float*)carve((size_t)96 * 4096 * sizeof(float));
    float*          ks      = (float*)carve((size_t)96 * 64 * sizeof(float));
    float*          zz      = (float*)carve((size_t)96 * 4096 * sizeof(float));

    pack_bias_kernel<<<9, 256, 0, stream>>>(q_bias, v_bias, qkvbias);

    // QKV: (32768 x 768) @ (768 x 2304) -> bf16
    gemm_wmma<true, true><<<(M / 128) * (2304 / 128), 256, 0, stream>>>(
        x, qkv_w, qkvbias, qkv, M, 2304, 768);

    // RoPE + softmax in place on q,k slices
    rope_softmax_kernel<<<(M * 12 * 2) / 8, 256, 0, stream>>>(qkv, rope);

    ksum_kernel<<<96, 256, 0, stream>>>(qkv, ks);
    z_kernel<<<(M * 12) / 8, 256, 0, stream>>>(qkv, ks, zz);
    kv_kernel<<<96, 128, 0, stream>>>(qkv, kv);
    attn_out_kernel<<<96 * 32, 256, 0, stream>>>(qkv, kv, zz, O);

    ln_kernel<<<M, 256, 0, stream>>>(O, norm_g, norm_b);

    // Projection: (32768 x 768) @ (768 x 768) -> f32 d_out
    gemm_wmma<false, false><<<(M / 128) * (768 / 128), 256, 0, stream>>>(
        O, proj_w, proj_b, out, M, 768, 768);
}
